// Baseline_Jinkyu_72713796321785
// MI455X (gfx1250) — matrix operand order repaired and checked by reading, not compiled, not run on hardware
//
#include <hip/hip_runtime.h>
#include <hip/hip_bf16.h>
#include <cstdint>
#include <cstddef>

// Problem dims (match reference)
#define B_   8
#define T_   100
#define L_   240
#define D_   256
#define H_   512
#define G4H  2048           // 4*H
#define LDW  (L_ * D_)      // 61440, row stride of W_ih

typedef __attribute__((ext_vector_type(16))) __bf16 v16bf;
typedef __attribute__((ext_vector_type(8)))  float  v8f;
typedef unsigned short u16t;
typedef __attribute__((ext_vector_type(8)))  u16t   v8us;
typedef __attribute__((ext_vector_type(4)))  unsigned int v4u;
typedef __attribute__((ext_vector_type(8)))  int    v8i;
typedef __attribute__((ext_vector_type(4)))  int    v4i;

struct U16x16 { v8us lo, hi; };   // 32 bytes == v16bf

#if __has_builtin(__builtin_amdgcn_tensor_load_to_lds)
#define HAVE_TDM 1
#else
#define HAVE_TDM 0
#endif

// ---- bf16 bit helpers (RNE), avoid relying on native __bf16 arithmetic ----
__device__ __forceinline__ u16t f2bf_bits(float f) {
  uint32_t u = __float_as_uint(f);
  uint32_t r = u + 0x7FFFu + ((u >> 16) & 1u);
  return (u16t)(r >> 16);
}
__device__ __forceinline__ float bf_bits2f(u16t b) {
  return __uint_as_float(((uint32_t)b) << 16);
}
__device__ __forceinline__ float sigmoidf_(float x) { return 1.0f / (1.0f + __expf(-x)); }

// ---- WMMA fragment loads (contiguous k = 16h + s, same order for A and B) ----
// B fragment (32x16 KxN), source stored row-major NxK: lane n = lane&15,
// K = h + (lane>=16 ? 16 : 0) -> 16 contiguous bf16 per lane.
__device__ __forceinline__ v16bf load_fragB(const u16t* p) {
  U16x16 t;
  t.lo = *(const v8us*)p;
  t.hi = *(const v8us*)(p + 8);
  return __builtin_bit_cast(v16bf, t);
}
// A fragment (16x32 MxK), row-major M rows: lane m = lane&15,
// K = h + (lane>=16?16:0) -> 16 contiguous bf16 per lane.
__device__ __forceinline__ v16bf load_fragA(const u16t* row, int k0, int lane) {
  int off = k0 + ((lane & 16) ? 16 : 0);
  U16x16 t;
  t.lo = *(const v8us*)(row + off);
  t.hi = *(const v8us*)(row + off + 8);
  return __builtin_bit_cast(v16bf, t);
}

#if HAVE_TDM
// ---- Tensor Data Mover: async 2D tile (512 rows x 32 cols of bf16) -> LDS ----
// D# group0: count=1 | lds_addr | global_addr[56:0] | type=2
// D# group1: data_size=1(2B); tensor_dim0/1 = 1<<30 (never OOB);
//            tile_dim0=32, tile_dim1=512, tile_dim2=0; dim0 stride = 61440.
__device__ __forceinline__ void tdm_load_b_slab(const u16t* gsrc, unsigned int lds_off) {
  unsigned long long ga = (unsigned long long)(uintptr_t)gsrc;
  v4u g0 = { 1u,                                   // count=1, user descriptor
             lds_off,                              // lds_addr (bytes)
             (unsigned int)(ga & 0xFFFFFFFFull),   // global_addr[31:0]
             (unsigned int)(((ga >> 32) & 0x01FFFFFFull) | (2u << 30)) }; // [56:32] | type=2
  v8i g1 = { (int)(1u << 16),                      // wg_mask=0, data_size=1 (2 bytes)
             0,                                    // abar=0 | tensor_dim0.lo16=0
             (int)0x00004000,                      // tensor_dim0.hi16 (1<<30) | tensor_dim1.lo16=0
             (int)(0x4000u | (32u << 16)),         // tensor_dim1.hi16 | tile_dim0=32
             (int)512,                             // tile_dim1=512 | tile_dim2=0
             (int)LDW,                             // tensor_dim0_stride[31:0] = 61440
             0, 0 };                               // stride0.hi16=0 | stride1=0
  v4i g2 = { 0, 0, 0, 0 };
  v4i g3 = { 0, 0, 0, 0 };
#if defined(__clang_major__) && (__clang_major__ >= 23)
  v8i g4 = { 0, 0, 0, 0, 0, 0, 0, 0 };
  __builtin_amdgcn_tensor_load_to_lds(g0, g1, g2, g3, g4, 0);
#else
  __builtin_amdgcn_tensor_load_to_lds(g0, g1, g2, g3, 0);
#endif
}
#endif // HAVE_TDM

// ============================ utility kernels ============================
__global__ void k_cvt_bf16(const float* __restrict__ in, u16t* __restrict__ out, int n) {
  int i = blockIdx.x * blockDim.x + threadIdx.x;
  if (i < n) out[i] = f2bf_bits(in[i]);
}
__global__ void k_zero(float* __restrict__ p, int n) {
  int i = blockIdx.x * blockDim.x + threadIdx.x;
  if (i < n) p[i] = 0.0f;
}

// ============================ proj GEMM (WMMA bf16) ============================
// proj[t,b,l,e] = sum_d cam[b,t,l,d]*pw_w[e,d] + pw_b[e]
// A = cam_bf16 (192000 x 256), B = pw_bf16 row-major [e][d] (N x K); one wave -> 16x64 tile.
// pw_w is only 128 KB bf16: stays L2/L0 resident, direct global fragment loads are fine.
__global__ void k_proj_gemm(const u16t* __restrict__ camb, const u16t* __restrict__ pwb,
                            const float* __restrict__ pw_bias, float* __restrict__ proj) {
  const int lane = threadIdx.x;        // 32 threads = 1 wave
  const int mt   = blockIdx.x;         // 0..11999
  const int nb   = blockIdx.y;         // 0..3  (cols nb*64)
  const int nlane = lane & 15;
  const int kb    = (lane & 16) ? 16 : 0;
  const u16t* arow = camb + (size_t)(mt * 16 + nlane) * D_;

  v8f acc[4] = {};
  for (int k0 = 0; k0 < D_; k0 += 32) {
    v16bf a = load_fragA(arow, k0, lane);
#pragma unroll
    for (int s = 0; s < 4; ++s) {
      int n = nb * 64 + s * 16 + nlane;
      v16bf bm = load_fragB(pwb + (size_t)n * D_ + k0 + kb);
      acc[s] = __builtin_amdgcn_wmma_f32_16x16x32_bf16(false, a, false, bm,
                                                       (short)0, acc[s], false, false);
    }
  }
  // (B,T,L) row tile -> (T,B,L) output rows; L%16==0 so a tile shares (b,t).
  int row0 = mt * 16;
  int b  = row0 / (T_ * L_);
  int tl = row0 % (T_ * L_);
  int t  = tl / L_;
  int l0 = tl % L_;
  size_t obase = ((size_t)t * B_ + b) * L_;
  const int moff = (lane & 16) ? 8 : 0;
#pragma unroll
  for (int v = 0; v < 8; ++v) {
    size_t orow = (obase + (size_t)(l0 + v + moff)) * D_;
#pragma unroll
    for (int s = 0; s < 4; ++s) {
      int col = nb * 64 + s * 16 + nlane;
      proj[orow + col] = acc[s][v] + pw_bias[col];
    }
  }
}

// ============================ G GEMM (WMMA bf16 + TDM double buffer) ============================
// G[t,b,l,j] = sum_d cam[b,t,l,d] * W_ih[j, l*D+d]   (240 GEMMs of (800x256)x(256x2048))
// Workgroup = 4 waves: 16 rows x 512 cols of one l. B slab (512x32 bf16 = 32KB) staged to LDS
// by the Tensor Data Mover, double-buffered so DMA(k+32) overlaps WMMA(k).
__global__ void __launch_bounds__(128)
k_g_gemm(const u16t* __restrict__ camb, const u16t* __restrict__ wihb,
         u16t* __restrict__ G) {
  const int tid  = threadIdx.x;        // 128 threads = 4 waves
  const int wave = tid >> 5;
  const int lane = tid & 31;
  const int mt = blockIdx.x;           // 0..49  rows over B*T = 800
  const int nB = blockIdx.y;           // 0..3   cols nB*512
  const int l  = blockIdx.z;           // 0..239
  const int nlane = lane & 15;
  const int kb    = (lane & 16) ? 16 : 0;
  const int r = mt * 16 + nlane;
  const int b = r / T_, t = r % T_;
  const u16t* arow = camb + (((size_t)(b * T_ + t)) * L_ + l) * D_;
  const int n0 = nB * 512;
  const u16t* wbase = wihb + (size_t)n0 * LDW + (size_t)l * D_;  // row n0, col l*D

  __shared__ __align__(128) u16t bslab[2][512 * 32];   // 2 x 32 KB double buffer

#if HAVE_TDM
  if (wave == 0)
    tdm_load_b_slab(wbase, (unsigned int)(uintptr_t)&bslab[0][0]);
#endif

  v8f acc[8] = {};
  for (int k0 = 0; k0 < D_; k0 += 32) {
    const int ib = (k0 >> 5) & 1;
#if HAVE_TDM
    if (wave == 0) {
      if (k0 + 32 < D_) {
        tdm_load_b_slab(wbase + (k0 + 32), (unsigned int)(uintptr_t)&bslab[ib ^ 1][0]);
        __builtin_amdgcn_s_wait_tensorcnt(1);   // slab(k0) done; slab(k0+32) in flight
      } else {
        __builtin_amdgcn_s_wait_tensorcnt(0);
      }
    }
    __syncthreads();                            // slab(k0) visible to all waves
#else
    // Fallback: cooperative global->LDS copy of the 512x32 slab.
    for (int e = tid; e < 512 * 2; e += 128) {  // 1024 chunks of 16 u16t
      int row = e >> 1, half = (e & 1) * 16;
      *(v8us*)&bslab[ib][row * 32 + half]     = *(const v8us*)(wbase + (size_t)row * LDW + k0 + half);
      *(v8us*)&bslab[ib][row * 32 + half + 8] = *(const v8us*)(wbase + (size_t)row * LDW + k0 + half + 8);
    }
    __syncthreads();
#endif
    v16bf a = load_fragA(arow, k0, lane);
    const u16t* sb = &bslab[ib][(wave * 128 + nlane) * 32 + kb];
#pragma unroll
    for (int s = 0; s < 8; ++s) {
      v16bf bm = load_fragB(sb + s * 16 * 32);  // ds_load_b128 x2
      acc[s] = __builtin_amdgcn_wmma_f32_16x16x32_bf16(false, a, false, bm,
                                                       (short)0, acc[s], false, false);
    }
    __syncthreads();                            // buffer reuse guard
  }

  const int moff = (lane & 16) ? 8 : 0;
#pragma unroll
  for (int v = 0; v < 8; ++v) {
    int rv = mt * 16 + v + moff;
    int bv = rv / T_, tv = rv % T_;
    size_t orow = (((size_t)(tv * B_ + bv)) * L_ + l) * (size_t)G4H;
#pragma unroll
    for (int s = 0; s < 8; ++s) {
      int col = n0 + wave * 128 + s * 16 + nlane;
      G[orow + col] = f2bf_bits(acc[s][v]);
    }
  }
}

// ============================ per-step attention ============================
// ha = hx @ w_w.T + w_b; s[l] = wa . tanh(proj[t,b,l,:] + ha) + wa_b; alpha = softmax_l(s)
__global__ void k_attn(const float* __restrict__ proj, const float* __restrict__ hx,
                       const float* __restrict__ w_w, const float* __restrict__ w_b,
                       const float* __restrict__ wa_w, const float* __restrict__ wa_b,
                       float* __restrict__ alpha, float* __restrict__ alpha_out, int t) {
  const int b = blockIdx.x;
  const int tid = threadIdx.x;   // 256
  __shared__ float hxs[H_], ha[D_], was[D_], red[256];
  hxs[tid]       = hx[b * H_ + tid];
  hxs[tid + 256] = hx[b * H_ + tid + 256];
  was[tid]       = wa_w[tid];
  __syncthreads();
  {
    float acc = w_b[tid];
    const float* wr = w_w + (size_t)tid * H_;
    for (int h = 0; h < H_; ++h) acc += hxs[h] * wr[h];
    ha[tid] = acc;
  }
  __syncthreads();
  float s = -3.0e38f;
  if (tid < L_) {
    const float* pr = proj + (((size_t)t * B_ + b) * L_ + tid) * D_;
    float sum = wa_b[0];
    for (int d = 0; d < D_; ++d) sum += tanhf(pr[d] + ha[d]) * was[d];
    s = sum;
  }
  red[tid] = s;
  for (int st = 128; st > 0; st >>= 1) { __syncthreads(); if (tid < st) red[tid] = fmaxf(red[tid], red[tid + st]); }
  __syncthreads();
  float mx = red[0];
  __syncthreads();
  float e = (tid < L_) ? __expf(s - mx) : 0.0f;
  red[tid] = e;
  for (int st = 128; st > 0; st >>= 1) { __syncthreads(); if (tid < st) red[tid] += red[tid + st]; }
  __syncthreads();
  float inv = 1.0f / red[0];
  if (tid < L_) {
    float a = e * inv;
    alpha[b * L_ + tid] = a;
    alpha_out[((size_t)t * B_ + b) * L_ + tid] = a;
  }
}

// ============================ per-step gates ============================
// gates[b,j] = b_ih[j]+b_hh[j] + sum_l alpha[b,l]*G[t,b,l,j] + sum_h hx[b,h]*W_hh[j,h]
__global__ void k_gates(const u16t* __restrict__ G, const float* __restrict__ alpha,
                        const float* __restrict__ hx, const float* __restrict__ W_hh,
                        const float* __restrict__ b_ih, const float* __restrict__ b_hh,
                        float* __restrict__ gates, int t) {
  const int b = blockIdx.x;
  const int tid = threadIdx.x;                 // 256
  const int j = blockIdx.y * 256 + tid;        // 0..2047
  __shared__ float al[L_], hxs[H_];
  if (tid < L_) al[tid] = alpha[b * L_ + tid];
  hxs[tid]       = hx[b * H_ + tid];
  hxs[tid + 256] = hx[b * H_ + tid + 256];
  __syncthreads();
  float acc = b_ih[j] + b_hh[j];
  const u16t* gp = G + (((size_t)t * B_ + b) * L_) * (size_t)G4H + j;
  for (int l = 0; l < L_; ++l) acc += al[l] * bf_bits2f(gp[(size_t)l * G4H]);
  const float* wr = W_hh + (size_t)j * H_;
  for (int h = 0; h < H_; ++h) acc += hxs[h] * wr[h];
  gates[b * G4H + j] = acc;
}

// ============================ per-step LSTM update + MLP head ============================
__global__ void k_lstm_mlp(const float* __restrict__ gates, float* __restrict__ hx, float* __restrict__ cx,
                           const float* __restrict__ m1w, const float* __restrict__ m1b,
                           const float* __restrict__ m2w, const float* __restrict__ m2b,
                           const float* __restrict__ m3w, const float* __restrict__ m3b,
                           const float* __restrict__ m4w, const float* __restrict__ m4b,
                           float* __restrict__ preds, int t) {
  const int b = blockIdx.x;
  const int j = threadIdx.x;   // 512
  __shared__ float hs[H_], x1[100], x2[50], x3[10];
  const float* g = gates + b * G4H;
  float gi = g[j], gf = g[H_ + j], gg = g[2 * H_ + j], go = g[3 * H_ + j];
  float c = sigmoidf_(gf) * cx[b * H_ + j] + sigmoidf_(gi) * tanhf(gg);
  float h = sigmoidf_(go) * tanhf(c);
  cx[b * H_ + j] = c;
  hx[b * H_ + j] = h;
  hs[j] = h;
  __syncthreads();
  if (j < 100) { float a = m1b[j]; const float* w = m1w + j * H_;  for (int k = 0; k < H_;  ++k) a += hs[k] * w[k]; x1[j] = fmaxf(a, 0.0f); }
  __syncthreads();
  if (j < 50)  { float a = m2b[j]; const float* w = m2w + j * 100; for (int k = 0; k < 100; ++k) a += x1[k] * w[k]; x2[j] = fmaxf(a, 0.0f); }
  __syncthreads();
  if (j < 10)  { float a = m3b[j]; const float* w = m3w + j * 50;  for (int k = 0; k < 50;  ++k) a += x2[k] * w[k]; x3[j] = fmaxf(a, 0.0f); }
  __syncthreads();
  if (j < 2)   { float a = m4b[j]; const float* w = m4w + j * 10;  for (int k = 0; k < 10;  ++k) a += x3[k] * w[k];
                 preds[((size_t)t * B_ + b) * 2 + j] = a; }
}

// ============================ host ============================
extern "C" void kernel_launch(void* const* d_in, const int* in_sizes, int n_in,
                              void* d_out, int out_size, void* d_ws, size_t ws_size,
                              hipStream_t stream) {
  (void)in_sizes; (void)n_in; (void)out_size; (void)ws_size;
  const float* cam  = (const float*)d_in[0];
  const float* pw_w = (const float*)d_in[1];
  const float* pw_b = (const float*)d_in[2];
  const float* w_w  = (const float*)d_in[3];
  const float* w_b  = (const float*)d_in[4];
  const float* wa_w = (const float*)d_in[5];
  const float* wa_b = (const float*)d_in[6];
  const float* W_ih = (const float*)d_in[7];
  const float* W_hh = (const float*)d_in[8];
  const float* b_ih = (const float*)d_in[9];
  const float* b_hh = (const float*)d_in[10];
  const float* m1w  = (const float*)d_in[11];
  const float* m1b  = (const float*)d_in[12];
  const float* m2w  = (const float*)d_in[13];
  const float* m2b  = (const float*)d_in[14];
  const float* m3w  = (const float*)d_in[15];
  const float* m3b  = (const float*)d_in[16];
  const float* m4w  = (const float*)d_in[17];
  const float* m4b  = (const float*)d_in[18];

  // workspace carve (aligned 256B): ~1.34 GB total
  char* ws = (char*)d_ws;
  size_t off = 0;
  auto carve = [&](size_t bytes) -> char* {
    char* p = ws + off;
    off = (off + bytes + 255) & ~(size_t)255;
    return p;
  };
  const size_t NCAM = (size_t)B_ * T_ * L_ * D_;        // 49,152,000
  const size_t NWIH = (size_t)G4H * LDW;                // 125,829,120
  u16t*  camb  = (u16t*)carve(NCAM * 2);
  u16t*  wihb  = (u16t*)carve(NWIH * 2);
  u16t*  pwb   = (u16t*)carve((size_t)D_ * D_ * 2);
  float* proj  = (float*)carve(NCAM * 4);               // (T,B,L,D)
  u16t*  G     = (u16t*)carve((size_t)T_ * B_ * L_ * G4H * 2); // (T,B,L,4H)
  float* hx    = (float*)carve((size_t)B_ * H_ * 4);
  float* cx    = (float*)carve((size_t)B_ * H_ * 4);
  float* alpha = (float*)carve((size_t)B_ * L_ * 4);
  float* gates = (float*)carve((size_t)B_ * G4H * 4);

  float* preds_out = (float*)d_out;                      // (T,B,2)
  float* alpha_out = (float*)d_out + (size_t)T_ * B_ * 2;// (T,B,L)

  // 1) one-time bf16 conversions (W_ih read from HBM exactly once)
  k_cvt_bf16<<<(int)((NCAM + 255) / 256), 256, 0, stream>>>(cam,  camb, (int)NCAM);
  k_cvt_bf16<<<(int)((NWIH + 255) / 256), 256, 0, stream>>>(W_ih, wihb, (int)NWIH);
  k_cvt_bf16<<<(D_ * D_ + 255) / 256,     256, 0, stream>>>(pw_w, pwb,  D_ * D_);

  // 2) big parallel WMMA GEMMs
  k_proj_gemm<<<dim3((B_ * T_ * L_) / 16, D_ / 64), 32, 0, stream>>>(camb, pwb, pw_b, proj);
  k_g_gemm<<<dim3((B_ * T_) / 16, G4H / 512, L_), 128, 0, stream>>>(camb, wihb, G);

  // 3) init recurrent state
  k_zero<<<(B_ * H_ + 255) / 256, 256, 0, stream>>>(hx, B_ * H_);
  k_zero<<<(B_ * H_ + 255) / 256, 256, 0, stream>>>(cx, B_ * H_);

  // 4) sequential recurrence: cheap per-step kernels (G reduce instead of W_ih stream)
  for (int t = 0; t < T_; ++t) {
    k_attn<<<B_, 256, 0, stream>>>(proj, hx, w_w, w_b, wa_w, wa_b, alpha, alpha_out, t);
    k_gates<<<dim3(B_, G4H / 256), 256, 0, stream>>>(G, alpha, hx, W_hh, b_ih, b_hh, gates, t);
    k_lstm_mlp<<<B_, H_, 0, stream>>>(gates, hx, cx, m1w, m1b, m2w, m2b, m3w, m3b, m4w, m4b,
                                      preds_out, t);
  }
}
